// GraphNCA_69441031242491
// MI455X (gfx1250) — matrix operand order repaired and checked by reading, not compiled, not run on hardware
//
#include <hip/hip_runtime.h>
#include <hip/hip_bf16.h>

// ---------------- CDNA5 / gfx1250 GraphNCA implementation -----------------
// wave32; WMMA f32_16x16x32_f16 for all convs + edge MLP; pixel-major
// ([pos][cin]) layouts so A/B fragments load as b128; TDM stages conv
// weights into LDS (TENSORcnt path).

typedef __attribute__((ext_vector_type(16))) _Float16 v16h;
typedef __attribute__((ext_vector_type(8)))  _Float16 v8h;
typedef __attribute__((ext_vector_type(8)))  float    v8f;
typedef __attribute__((ext_vector_type(4)))  unsigned int u32x4;
typedef __attribute__((ext_vector_type(8)))  int      i32x8;
typedef __attribute__((ext_vector_type(4)))  int      i32x4;

#define CCH   16          // C
#define DQK   32          // D
#define HIDN  64          // HID
#define NNODE 64          // P*P
#define BB    16          // B
#define SP    256         // HP*WP
#define IMG   1024        // B*N
#define EE    224         // directed edges of 8x8 4-neighborhood grid
#define KIT   4

#if __has_builtin(__builtin_amdgcn_tensor_load_to_lds) && \
    __has_builtin(__builtin_amdgcn_s_wait_tensorcnt)
#define HAVE_TDM 1
#else
#define HAVE_TDM 0
#endif

__device__ __forceinline__ v16h cat16(v8h lo, v8h hi) {
    return __builtin_shufflevector(lo, hi, 0, 1, 2, 3, 4, 5, 6, 7,
                                   8, 9, 10, 11, 12, 13, 14, 15);
}
__device__ __forceinline__ float sigm(float x) {
    return 1.f / (1.f + __expf(-x));
}

#if HAVE_TDM
// 1D (one-row 2D tile) TDM copy: n_half f16 elements global -> LDS.
// D# per CDNA5 ISA ch.8: group0 {count=1, lds_addr, global_addr, type=2},
// group1 {data_size=2B, tensor_dim0=n, tensor_dim1=1, tile_dim0=n,
// tile_dim1=1, stride0=n}.
__device__ __forceinline__ void tdm_load_1d_f16(const _Float16* gsrc,
                                                unsigned lds_off, int n_half)
{
    unsigned long long ga = (unsigned long long)(size_t)gsrc;
    u32x4 g0;
    g0[0] = 1u;                                   // count=1
    g0[1] = lds_off;                              // lds_addr
    g0[2] = (unsigned)ga;                         // global_addr[31:0]
    g0[3] = (unsigned)((ga >> 32) & 0x01FFFFFFull) | (2u << 30); // [56:32]|type
    i32x8 g1;
    g1[0] = (1 << 16);                            // data_size = 2 bytes
    g1[1] = (int)((n_half & 0xFFFF) << 16);       // tensor_dim0 lo
    g1[2] = (int)((n_half >> 16) & 0xFFFF) | (1 << 16); // dim0 hi | dim1 lo=1
    g1[3] = (int)((n_half & 0xFFFF) << 16);       // tile_dim0
    g1[4] = 1;                                    // tile_dim1 = 1
    g1[5] = n_half;                               // tensor_dim0_stride lo
    g1[6] = 0; g1[7] = 0;
    i32x4 gz; gz[0] = gz[1] = gz[2] = gz[3] = 0;
#if defined(__clang_major__) && __clang_major__ >= 23
    i32x8 gz8; gz8[0]=gz8[1]=gz8[2]=gz8[3]=gz8[4]=gz8[5]=gz8[6]=gz8[7]=0;
    __builtin_amdgcn_tensor_load_to_lds(g0, g1, gz, gz, gz8, 0);
#else
    __builtin_amdgcn_tensor_load_to_lds(g0, g1, gz, gz, 0);
#endif
}
#endif

// ---------------------------------------------------------------- prep ----
// f16 weight matrices in [Cout][K] layout; T and MLP weights converted.
__global__ __launch_bounds__(256) void prep_kernel(
    const float* __restrict__ wm, const float* __restrict__ wg,
    const float* __restrict__ wc, const float* __restrict__ g_w,
    const float* __restrict__ aw1, const float* __restrict__ aw2,
    _Float16* __restrict__ wm16, _Float16* __restrict__ wg16,
    _Float16* __restrict__ wc16, _Float16* __restrict__ w1h,
    _Float16* __restrict__ w2h, float* __restrict__ T)
{
    int idx = blockIdx.x * 256 + threadIdx.x;
    if (idx < 16 * 160) {                 // wm16[n][k], K padded 144->160
        int n = idx / 160, kk = idx % 160;
        float v = 0.f;
        if (kk < 144) {
            int tap = kk >> 4, cin = kk & 15;
            v = wm[(n * 16 + cin) * 9 + tap];
        }
        wm16[idx] = (_Float16)v;
        return;
    }
    int i2 = idx - 16 * 160;
    if (i2 < 32 * 288) {                  // wg16[n][k]
        int n = i2 / 288, kk = i2 % 288;
        int tap = kk >> 5, cin = kk & 31;
        wg16[i2] = (_Float16)wg[(n * 32 + cin) * 9 + tap];
        return;
    }
    int i3 = i2 - 32 * 288;
    if (i3 < 16 * 288) {                  // wc16[n][k]
        int n = i3 / 288, kk = i3 % 288;
        int tap = kk >> 5, cin = kk & 31;
        wc16[i3] = (_Float16)wc[(n * 32 + cin) * 9 + tap];
        return;
    }
    int i4 = i3 - 16 * 288;
    if (i4 < HIDN * HIDN) {               // w1h[o][i] = aw1[i][o]
        int o = i4 >> 6, i = i4 & 63;
        w1h[i4] = (_Float16)aw1[i * HIDN + o];
        return;
    }
    int i5 = i4 - HIDN * HIDN;
    if (i5 < HIDN * HIDN) {
        int o = i5 >> 6, i = i5 & 63;
        w2h[i5] = (_Float16)aw2[i * HIDN + o];
        return;
    }
    int i6 = i5 - HIDN * HIDN;
    if (i6 < SP * CCH) {                  // T[pos][c]
        int pos = i6 >> 4, c = i6 & 15, y = pos >> 4, x = pos & 15;
        float s = 0.f;
        for (int dy = 0; dy < 3; ++dy)
            for (int dx = 0; dx < 3; ++dx) {
                int yy = y + dy - 1, xx = x + dx - 1;
                if (yy >= 0 && yy < 16 && xx >= 0 && xx < 16)
                    s += g_w[c * 9 + dy * 3 + dx];
            }
        T[i6] = s;
    }
}

// ------------------------------------------------------------ init / out --
// X global layout: [img][pos][c16] (pixel-major).
__global__ __launch_bounds__(256) void init_x_kernel(
    const float* __restrict__ seed, float* __restrict__ X)
{
    int idx = blockIdx.x * 256 + threadIdx.x;            // IMG*256*16
    int c = idx & 15, pos = (idx >> 4) & 255, img = idx >> 12;
    int b = img >> 6, node = img & 63, py = node >> 3, px = node & 7;
    int y = pos >> 4, x = pos & 15;
    X[idx] = seed[((b * 16 + c) * 128 + py * 16 + y) * 128 + px * 16 + x];
}

__global__ __launch_bounds__(256) void out_kernel(
    const float* __restrict__ X, float* __restrict__ out)
{
    int idx = blockIdx.x * 256 + threadIdx.x;            // B*C*128*128
    int x = idx & 127, y = (idx >> 7) & 127, c = (idx >> 14) & 15, b = idx >> 18;
    int py = y >> 4, px = x >> 4, yy = y & 15, xx = x & 15;
    out[idx] = X[(((b * 64 + py * 8 + px) * 256) + yy * 16 + xx) * 16 + c];
}

// ------------------------------------------------------------- mean/q/k ---
__global__ __launch_bounds__(256) void mean_qk_kernel(
    const float* __restrict__ X,
    const float* __restrict__ wq, const float* __restrict__ bq,
    const float* __restrict__ wk, const float* __restrict__ bk,
    float* __restrict__ qv, float* __restrict__ kv)
{
    __shared__ float s_xm[CCH];
    int img = blockIdx.x, tid = threadIdx.x;
    int wave = tid >> 5, lane = tid & 31;
    const float* xin = X + (size_t)img * SP * CCH;
    for (int cc = 0; cc < 2; ++cc) {
        int c = wave * 2 + cc;
        float s = 0.f;
        for (int i = lane; i < SP; i += 32) s += xin[i * CCH + c];
        for (int off = 16; off; off >>= 1) s += __shfl_xor(s, off, 32);
        if (lane == 0) s_xm[c] = s * (1.f / 256.f);
    }
    __syncthreads();
    if (tid < DQK) {
        int d = tid;
        float sq = bq[d], sk = bk[d];
        for (int c = 0; c < CCH; ++c) {
            float xm = s_xm[c];
            sq += xm * wq[c * DQK + d];
            sk += xm * wk[c * DQK + d];
        }
        qv[img * DQK + d] = sq;
        kv[img * DQK + d] = sk;
    }
}

// ------------------------------------------------------------- edge MLP ---
__global__ __launch_bounds__(128) void edge_mlp_kernel(
    const float* __restrict__ qv, const float* __restrict__ kv,
    const int* __restrict__ edge_index,
    const _Float16* __restrict__ w1h, const float* __restrict__ b1,
    const _Float16* __restrict__ w2h, const float* __restrict__ b2,
    const float* __restrict__ wo, const float* __restrict__ bo,
    float* __restrict__ evals, int rows)
{
    __shared__ __align__(16) _Float16 h0[16][HIDN];
    __shared__ __align__(16) _Float16 h1[16][HIDN];
    int row0 = blockIdx.x * 16, tid = threadIdx.x;

    for (int i = tid; i < 16 * HIDN; i += 128) {
        int r = i >> 6, c = i & 63;
        int g = row0 + r;
        float v = 0.f;
        if (g < rows) {
            int b = g / EE, e = g % EE;
            int ei = edge_index[2 * e + 0], ej = edge_index[2 * e + 1];
            v = (c < DQK) ? qv[(b * NNODE + ei) * DQK + c]
                          : kv[(b * NNODE + ej) * DQK + (c - DQK)];
        }
        h0[r][c] = (_Float16)v;
    }
    __syncthreads();

    int wave = tid >> 5, lane = tid & 31;
    int n0 = wave * 16;
    int am = lane & 15, akh = (lane >> 4) << 3;
    int bkb = (lane >> 4) << 3, bn = lane & 15;

    // ---- layer 1 ----
    v8f acc;
#pragma unroll
    for (int v = 0; v < 8; ++v) acc[v] = 0.f;
#pragma unroll
    for (int kc = 0; kc < HIDN; kc += 32) {
        const v8h* pa = (const v8h*)&h0[am][kc + akh];
        v16h a = cat16(pa[0], pa[2]);
        const v8h* pb = (const v8h*)(w1h + (n0 + bn) * HIDN + kc + bkb);
        v16h bf = cat16(pb[0], pb[2]);
        acc = __builtin_amdgcn_wmma_f32_16x16x32_f16(false, a, false, bf,
                                                     (short)0, acc, false, false);
    }
#pragma unroll
    for (int v = 0; v < 8; ++v) {
        int r = v + ((lane >> 4) << 3);
        float t = acc[v] + b1[n0 + bn];
        h1[r][n0 + bn] = (_Float16)(t > 0.f ? t : 0.f);
    }
    __syncthreads();

    // ---- layer 2 (write back into h0) ----
#pragma unroll
    for (int v = 0; v < 8; ++v) acc[v] = 0.f;
#pragma unroll
    for (int kc = 0; kc < HIDN; kc += 32) {
        const v8h* pa = (const v8h*)&h1[am][kc + akh];
        v16h a = cat16(pa[0], pa[2]);
        const v8h* pb = (const v8h*)(w2h + (n0 + bn) * HIDN + kc + bkb);
        v16h bf = cat16(pb[0], pb[2]);
        acc = __builtin_amdgcn_wmma_f32_16x16x32_f16(false, a, false, bf,
                                                     (short)0, acc, false, false);
    }
#pragma unroll
    for (int v = 0; v < 8; ++v) {
        int r = v + ((lane >> 4) << 3);
        float t = acc[v] + b2[n0 + bn];
        h0[r][n0 + bn] = (_Float16)(t > 0.f ? t : 0.f);
    }
    __syncthreads();

    if (tid < 16) {
        int g = row0 + tid;
        if (g < rows) {
            float s = bo[0];
            for (int c = 0; c < HIDN; ++c) s += (float)h0[tid][c] * wo[c];
            evals[g] = s;
        }
    }
}

// -------------------------------------------------- conv_m (16->16, 3x3) --
// LDS input: [pix18x18][cin16] f16; output M16: [img][pos][cout16].
__global__ __launch_bounds__(256) void conv_m_kernel(
    const float* __restrict__ X, const _Float16* __restrict__ wm16,
    const float* __restrict__ bm, _Float16* __restrict__ M16)
{
    __shared__ __align__(16) _Float16 s_in[18 * 18 * CCH];
    __shared__ __align__(16) _Float16 s_wm[16 * 160];
    int img = blockIdx.x, tid = threadIdx.x;

#if HAVE_TDM
    if (tid < 32) {
        tdm_load_1d_f16(wm16, (unsigned)(size_t)(void*)s_wm, 16 * 160);
        __builtin_amdgcn_s_wait_tensorcnt(0);
    }
#else
    for (int i = tid * 8; i < 16 * 160; i += 256 * 8)
        *(v8h*)&s_wm[i] = *(const v8h*)&wm16[i];
#endif
    {   // zero-fill padded LDS tile (b128 stores)
        v8h z;
#pragma unroll
        for (int v = 0; v < 8; ++v) z[v] = (_Float16)0.f;
        for (int i = tid * 8; i < 18 * 18 * CCH; i += 256 * 8)
            *(v8h*)&s_in[i] = z;
    }
    __syncthreads();
    const float* xin = X + (size_t)img * SP * CCH;
    for (int i = tid; i < SP; i += 256) {                // one pixel/thread
        int y = i >> 4, x = i & 15;
        int pix = (y + 1) * 18 + (x + 1);
        const float* xp = xin + i * CCH;
        v8h h0v, h1v;
#pragma unroll
        for (int c = 0; c < 8; ++c) {
            h0v[c] = (_Float16)xp[c];
            h1v[c] = (_Float16)xp[c + 8];
        }
        *(v8h*)&s_in[pix * CCH + 0] = h0v;
        *(v8h*)&s_in[pix * CCH + 8] = h1v;
    }
    __syncthreads();

    int wave = tid >> 5, lane = tid & 31;
    int am = lane & 15, akh = (lane >> 4) << 3;
    int bkb = (lane >> 4) << 3, bn = lane & 15;
    _Float16* mout = M16 + (size_t)img * SP * CCH;

    for (int t = 0; t < 2; ++t) {
        int m0 = (wave * 2 + t) * 16;
        int pos = m0 + am, y = pos >> 4, x = pos & 15;
        v8f acc;
#pragma unroll
        for (int v = 0; v < 8; ++v) acc[v] = 0.f;
#pragma unroll
        for (int kc = 0; kc < 160; kc += 32) {
            int tapA = kc >> 4;                          // group A tap
            int tapB = tapA + 1; tapB = tapB > 8 ? 8 : tapB;  // pad clamp
            int pixA = (y + tapA / 3) * 18 + (x + tapA % 3);
            int pixB = (y + tapB / 3) * 18 + (x + tapB % 3);
            v8h lo = *(const v8h*)&s_in[pixA * CCH + akh];
            v8h hi = *(const v8h*)&s_in[pixB * CCH + akh];
            v16h a = cat16(lo, hi);
            const v8h* pb = (const v8h*)&s_wm[bn * 160 + kc + bkb];
            v16h bf = cat16(pb[0], pb[2]);
            acc = __builtin_amdgcn_wmma_f32_16x16x32_f16(false, a, false, bf,
                                                         (short)0, acc, false, false);
        }
#pragma unroll
        for (int v = 0; v < 8; ++v) {
            int p = m0 + v + ((lane >> 4) << 3);
            mout[p * CCH + bn] = (_Float16)(acc[v] + bm[bn]);
        }
    }
}

// ---------------------------------------------------- gated aggregation ---
// Deterministic in-order edge scan; pixel-major M/Magg/T -> b128 traffic.
__global__ __launch_bounds__(256) void agg_kernel(
    const _Float16* __restrict__ M16, const float* __restrict__ evals,
    const float* __restrict__ T, const float* __restrict__ g_b,
    const int* __restrict__ edge_index, _Float16* __restrict__ Magg16)
{
    __shared__ int   s_ei[EE], s_ej[EE];
    __shared__ float s_ev[EE];
    int b = blockIdx.x >> 6, node = blockIdx.x & 63;
    int tid = threadIdx.x;
    for (int e = tid; e < EE; e += 256) {
        s_ei[e] = edge_index[2 * e + 0];
        s_ej[e] = edge_index[2 * e + 1];
        s_ev[e] = evals[b * EE + e];
    }
    __syncthreads();
    int pos = tid;
    float tc[CCH], gb[CCH];
#pragma unroll
    for (int c = 0; c < CCH; ++c) { tc[c] = T[pos * CCH + c]; gb[c] = g_b[c]; }
    float acc[CCH];
#pragma unroll
    for (int c = 0; c < CCH; ++c) acc[c] = 0.f;
    for (int e = 0; e < EE; ++e) {
        if (s_ei[e] == node) {
            float ev = s_ev[e];
            const v8h* mp = (const v8h*)(M16 +
                ((size_t)(b * NNODE + s_ej[e]) * SP + pos) * CCH);
            v8h mlo = mp[0], mhi = mp[1];
#pragma unroll
            for (int c = 0; c < 8; ++c) {
                acc[c]     += sigm(ev * tc[c]     + gb[c])     * (float)mlo[c];
                acc[c + 8] += sigm(ev * tc[c + 8] + gb[c + 8]) * (float)mhi[c];
            }
        }
    }
    v8h o0, o1;
#pragma unroll
    for (int c = 0; c < 8; ++c) {
        o0[c] = (_Float16)acc[c];
        o1[c] = (_Float16)acc[c + 8];
    }
    _Float16* mout = Magg16 + ((size_t)blockIdx.x * SP + pos) * CCH;
    *(v8h*)&mout[0] = o0;
    *(v8h*)&mout[8] = o1;
}

// ---------------------------------------------- fused GRU (two WMMA convs)
__global__ __launch_bounds__(256) void gru_kernel(
    float* __restrict__ X, const _Float16* __restrict__ Magg16,
    const _Float16* __restrict__ wg16, const float* __restrict__ bg,
    const _Float16* __restrict__ wc16, const float* __restrict__ bc,
    float* __restrict__ Xout)
{
    __shared__ __align__(16) _Float16 s_in[18 * 18 * 32]; // [pix][Magg16|h16]
    __shared__ __align__(16) _Float16 s_z[SP * CCH];      // [pos][c]
    __shared__ __align__(16) _Float16 s_wg[32 * 288];
    __shared__ __align__(16) _Float16 s_wc[16 * 288];
    int img = blockIdx.x, tid = threadIdx.x;

#if HAVE_TDM
    if (tid < 32) {
        tdm_load_1d_f16(wg16, (unsigned)(size_t)(void*)s_wg, 32 * 288);
        tdm_load_1d_f16(wc16, (unsigned)(size_t)(void*)s_wc, 16 * 288);
        __builtin_amdgcn_s_wait_tensorcnt(0);
    }
#else
    for (int i = tid * 8; i < 32 * 288; i += 256 * 8)
        *(v8h*)&s_wg[i] = *(const v8h*)&wg16[i];
    for (int i = tid * 8; i < 16 * 288; i += 256 * 8)
        *(v8h*)&s_wc[i] = *(const v8h*)&wc16[i];
#endif
    {
        v8h z;
#pragma unroll
        for (int v = 0; v < 8; ++v) z[v] = (_Float16)0.f;
        for (int i = tid * 8; i < 18 * 18 * 32; i += 256 * 8)
            *(v8h*)&s_in[i] = z;
    }
    __syncthreads();
    const float*    xin  = X      + (size_t)img * SP * CCH;
    const _Float16* magg = Magg16 + (size_t)img * SP * CCH;
    for (int i = tid; i < SP; i += 256) {                // one pixel/thread
        int y = i >> 4, x = i & 15;
        int pix = (y + 1) * 18 + (x + 1);
        const v8h* mg = (const v8h*)(magg + i * CCH);
        *(v8h*)&s_in[pix * 32 + 0] = mg[0];
        *(v8h*)&s_in[pix * 32 + 8] = mg[1];
        const float* xp = xin + i * CCH;
        v8h h0v, h1v;
#pragma unroll
        for (int c = 0; c < 8; ++c) {
            h0v[c] = (_Float16)xp[c];
            h1v[c] = (_Float16)xp[c + 8];
        }
        *(v8h*)&s_in[pix * 32 + 16] = h0v;
        *(v8h*)&s_in[pix * 32 + 24] = h1v;
    }
    __syncthreads();

    int wave = tid >> 5, lane = tid & 31;
    int am = lane & 15, akh = (lane >> 4) << 3;
    int bkb = (lane >> 4) << 3, bn = lane & 15;

    // ---- gate conv: Cout=32, K=288 (chunk == tap, Cin=32) ----
    v8f accg[4];
#pragma unroll
    for (int j = 0; j < 4; ++j)
#pragma unroll
        for (int v = 0; v < 8; ++v) accg[j][v] = 0.f;
    for (int j = 0; j < 4; ++j) {
        int tile = wave + 8 * j;
        int rowtile = tile & 15, coltile = tile >> 4;
        int m0 = rowtile * 16;
        int pos = m0 + am, y = pos >> 4, x = pos & 15;
        int n = coltile * 16 + bn;
#pragma unroll
        for (int kc = 0; kc < 288; kc += 32) {
            int tap = kc >> 5;
            int pix = (y + tap / 3) * 18 + (x + tap % 3);
            v8h lo = *(const v8h*)&s_in[pix * 32 + akh];
            v8h hi = *(const v8h*)&s_in[pix * 32 + 16 + akh];
            v16h a = cat16(lo, hi);
            const v8h* pb = (const v8h*)&s_wg[n * 288 + kc + bkb];
            v16h bf = cat16(pb[0], pb[2]);
            accg[j] = __builtin_amdgcn_wmma_f32_16x16x32_f16(false, a, false, bf,
                                                             (short)0, accg[j], false, false);
        }
    }
    __syncthreads();                                    // all h reads done

    // sigmoid -> keep z, fold r into h channels of s_in
    for (int j = 0; j < 4; ++j) {
        int tile = wave + 8 * j;
        int rowtile = tile & 15, coltile = tile >> 4;
        int m0 = rowtile * 16;
        int cout = coltile * 16 + bn;
#pragma unroll
        for (int v = 0; v < 8; ++v) {
            int pos = m0 + v + ((lane >> 4) << 3);
            float g = sigm(accg[j][v] + bg[cout]);
            if (cout < 16) {
                s_z[pos * CCH + cout] = (_Float16)g;
            } else {
                int y = pos >> 4, x = pos & 15;
                int idx = ((y + 1) * 18 + (x + 1)) * 32 + cout; // h chan=cout
                s_in[idx] = (_Float16)((float)s_in[idx] * g);
            }
        }
    }
    __syncthreads();

    // ---- candidate conv: Cout=16, K=288 ----
    for (int t = 0; t < 2; ++t) {
        int m0 = (wave * 2 + t) * 16;
        int pos0 = m0 + am, y = pos0 >> 4, x = pos0 & 15;
        v8f acc;
#pragma unroll
        for (int v = 0; v < 8; ++v) acc[v] = 0.f;
#pragma unroll
        for (int kc = 0; kc < 288; kc += 32) {
            int tap = kc >> 5;
            int pix = (y + tap / 3) * 18 + (x + tap % 3);
            v8h lo = *(const v8h*)&s_in[pix * 32 + akh];
            v8h hi = *(const v8h*)&s_in[pix * 32 + 16 + akh];
            v16h a = cat16(lo, hi);
            const v8h* pb = (const v8h*)&s_wc[bn * 288 + kc + bkb];
            v16h bf = cat16(pb[0], pb[2]);
            acc = __builtin_amdgcn_wmma_f32_16x16x32_f16(false, a, false, bf,
                                                         (short)0, acc, false, false);
        }
        int cout = bn;
#pragma unroll
        for (int v = 0; v < 8; ++v) {
            int pos = m0 + v + ((lane >> 4) << 3);
            float cand = tanhf(acc[v] + bc[cout]);
            float z = (float)s_z[pos * CCH + cout];
            size_t xi = ((size_t)img * SP + pos) * CCH + cout;
            float h = X[xi];
            Xout[xi] = (1.f - z) * h + z * cand;        // own (cout,pos)
        }
    }
}

// ------------------------------------------------------------------ host --
extern "C" void kernel_launch(void* const* d_in, const int* in_sizes, int n_in,
                              void* d_out, int out_size, void* d_ws, size_t ws_size,
                              hipStream_t stream) {
    const float* seed       = (const float*)d_in[0];
    const int*   edge_index = (const int*)  d_in[1];
    const float* wq  = (const float*)d_in[2];
    const float* bq  = (const float*)d_in[3];
    const float* wk  = (const float*)d_in[4];
    const float* bk  = (const float*)d_in[5];
    const float* wm  = (const float*)d_in[6];
    const float* bm  = (const float*)d_in[7];
    const float* aw1 = (const float*)d_in[8];
    const float* ab1 = (const float*)d_in[9];
    const float* aw2 = (const float*)d_in[10];
    const float* ab2 = (const float*)d_in[11];
    const float* awo = (const float*)d_in[12];
    const float* abo = (const float*)d_in[13];
    const float* g_w = (const float*)d_in[14];
    const float* g_b = (const float*)d_in[15];
    const float* gwg = (const float*)d_in[16];
    const float* gbg = (const float*)d_in[17];
    const float* gwc = (const float*)d_in[18];
    const float* gbc = (const float*)d_in[19];

    // workspace carve (f32 first, then f16; 16B-aligned boundaries)
    float* X     = (float*)d_ws;                  // IMG*256*16
    float* qv    = X     + (size_t)IMG * SP * CCH;
    float* kv    = qv    + (size_t)IMG * DQK;
    float* evals = kv    + (size_t)IMG * DQK;
    float* T     = evals + (size_t)BB * EE;
    _Float16* M16    = (_Float16*)(T + SP * CCH);
    _Float16* Magg16 = M16    + (size_t)IMG * SP * CCH;
    _Float16* wm16   = Magg16 + (size_t)IMG * SP * CCH;
    _Float16* wg16   = wm16   + 16 * 160;
    _Float16* wc16   = wg16   + 32 * 288;
    _Float16* w1h    = wc16   + 16 * 288;
    _Float16* w2h    = w1h    + HIDN * HIDN;

    init_x_kernel<<<(IMG * SP * CCH) / 256, 256, 0, stream>>>(seed, X);
    {
        int total = 16 * 160 + 32 * 288 + 16 * 288 + 2 * HIDN * HIDN + SP * CCH;
        prep_kernel<<<(total + 255) / 256, 256, 0, stream>>>(
            wm, gwg, gwc, g_w, aw1, aw2, wm16, wg16, wc16, w1h, w2h, T);
    }

    for (int it = 0; it < KIT; ++it) {
        mean_qk_kernel<<<IMG, 256, 0, stream>>>(X, wq, bq, wk, bk, qv, kv);
        edge_mlp_kernel<<<(BB * EE) / 16, 128, 0, stream>>>(
            qv, kv, edge_index, w1h, ab1, w2h, ab2, awo, abo, evals, BB * EE);
        conv_m_kernel<<<IMG, 256, 0, stream>>>(X, wm16, bm, M16);
        agg_kernel<<<IMG, 256, 0, stream>>>(M16, evals, T, g_b, edge_index, Magg16);
        gru_kernel<<<IMG, 256, 0, stream>>>(X, Magg16, wg16, gbg, wc16, gbc, X);
    }

    out_kernel<<<(BB * CCH * 128 * 128) / 256, 256, 0, stream>>>(X, (float*)d_out);
}
